// SelfAttention_86449101733967
// MI455X (gfx1250) — matrix operand order repaired and checked by reading, not compiled, not run on hardware
//
#include <hip/hip_runtime.h>
#include <hip/hip_bf16.h>

#define B_  4
#define C_  128
#define N_  4096
#define CK_ 32

typedef _Float16 v8h  __attribute__((ext_vector_type(8)));
typedef _Float16 v16h __attribute__((ext_vector_type(16)));
typedef float    v8f  __attribute__((ext_vector_type(8)));

// Workspace layout (needs ~6.07 MB):
//   fT : [B][N][32] f16 @ 0      (1 MB)   f transposed, row n contiguous over k
//   gT : [B][N][32] f16 @ 1 MB   (1 MB)
//   vs : [B][C][N]  f16 @ 2 MB   (4 MB)   v projection, row c contiguous over n
//   sh : [B][N]     f32 @ 6 MB   (64 KB)  shift[n] = max_m s + log(sum_m exp(s-max))

// Build 16-bit WMMA A-operand half for this lane: elements 0..15 = base[0..15]
// (contiguous K grouping {16h..16h+15})
__device__ __forceinline__ v16h load_a16(const _Float16* __restrict__ base) {
  v8h lo = *(const v8h*)(base);
  v8h hi = *(const v8h*)(base + 8);
  return __builtin_shufflevector(lo, hi, 0,1,2,3,4,5,6,7,8,9,10,11,12,13,14,15);
}

// Two f32 -> one packed f16x2 dword (round-to-nearest-even per element).
__device__ __forceinline__ unsigned int packh2(float a, float b) {
  union { _Float16 h[2]; unsigned int u; } t;
  t.h[0] = (_Float16)a;
  t.h[1] = (_Float16)b;
  return t.u;
}

// ---------------- Projection kernels (f32 math, f16 outputs) ----------------
__global__ void proj_fg(const float* __restrict__ x,  const float* __restrict__ Wf,
                        const float* __restrict__ bf, const float* __restrict__ Wg,
                        const float* __restrict__ bg, _Float16* __restrict__ fT,
                        _Float16* __restrict__ gT) {
  int t = blockIdx.x * blockDim.x + threadIdx.x;   // B*N*CK threads, n fastest
  int n = t & (N_ - 1);
  int k = (t >> 12) & (CK_ - 1);
  int b = t >> 17;
  const float* xp = x + (size_t)b * C_ * N_ + n;
  const float* wf = Wf + k * C_;
  const float* wg = Wg + k * C_;
  float af = bf[k], ag = bg[k];
#pragma unroll 8
  for (int c = 0; c < C_; ++c) {
    float xv = xp[(size_t)c * N_];
    af += wf[c] * xv;
    ag += wg[c] * xv;
  }
  size_t o = ((size_t)(b * N_ + n)) * CK_ + k;
  fT[o] = (_Float16)af;
  gT[o] = (_Float16)ag;
}

__global__ void proj_v(const float* __restrict__ x, const float* __restrict__ Wh,
                       const float* __restrict__ bh, _Float16* __restrict__ vs) {
  int t = blockIdx.x * blockDim.x + threadIdx.x;   // B*C*N threads, n fastest
  int n = t & (N_ - 1);
  int c = (t >> 12) & (C_ - 1);
  int b = t >> 19;
  const float* xp = x + (size_t)b * C_ * N_ + n;
  const float* wh = Wh + c * C_;
  float a = bh[c];
#pragma unroll 8
  for (int c2 = 0; c2 < C_; ++c2) a += wh[c2] * xp[(size_t)c2 * N_];
  vs[((size_t)(b * C_ + c)) * N_ + n] = (_Float16)a;
}

// ---------------- Pass 1: softmax row statistics via WMMA -------------------
// One wave owns 16 rows n; loops all m tiles. shift = max + log(sumexp).
__global__ void __launch_bounds__(128) attn_stats(const _Float16* __restrict__ fT,
                                                  const _Float16* __restrict__ gT,
                                                  float* __restrict__ shift) {
  const int l   = threadIdx.x & 31;
  const int w   = threadIdx.x >> 5;
  const int r16 = l & 15;
  const int h   = l >> 4;
  const int b   = blockIdx.y;
  const int n0  = (blockIdx.x * 4 + w) * 16;

  // A operand: rows n0+r16, K = ck (32). Loaded once, reused for every m tile.
  const v16h a = load_a16(fT + ((size_t)(b * N_ + n0 + r16)) * CK_ + 16 * h);
  const _Float16* gb = gT + ((size_t)b * N_ + r16) * CK_ + 16 * h;

  float mx[8];
#pragma unroll
  for (int r = 0; r < 8; ++r) mx[r] = -3.0e38f;
  for (int mi = 0; mi < N_ / 16; ++mi) {
    v16h g = *(const v16h*)(gb + (size_t)mi * 16 * CK_);
    v8f z = {};
    v8f s = __builtin_amdgcn_wmma_f32_16x16x32_f16(false, a, false, g, (short)0, z, false, false);
#pragma unroll
    for (int r = 0; r < 8; ++r) mx[r] = fmaxf(mx[r], s[r]);
  }
#pragma unroll
  for (int d = 8; d >= 1; d >>= 1) {
#pragma unroll
    for (int r = 0; r < 8; ++r) mx[r] = fmaxf(mx[r], __shfl_xor(mx[r], d, 32));
  }

  float sm[8];
#pragma unroll
  for (int r = 0; r < 8; ++r) sm[r] = 0.0f;
  for (int mi = 0; mi < N_ / 16; ++mi) {
    v16h g = *(const v16h*)(gb + (size_t)mi * 16 * CK_);
    v8f z = {};
    v8f s = __builtin_amdgcn_wmma_f32_16x16x32_f16(false, a, false, g, (short)0, z, false, false);
#pragma unroll
    for (int r = 0; r < 8; ++r) sm[r] += __expf(s[r] - mx[r]);
  }
#pragma unroll
  for (int d = 8; d >= 1; d >>= 1) {
#pragma unroll
    for (int r = 0; r < 8; ++r) sm[r] += __shfl_xor(sm[r], d, 32);
  }
  if (r16 == 0) {
#pragma unroll
    for (int r = 0; r < 8; ++r)
      shift[b * N_ + n0 + 8 * h + r] = mx[r] + __logf(sm[r]);
  }
}

// ---------------- Pass 2: fused s-recompute + exp + v@p ---------------------
// One wave owns one m-tile (16 cols) and all 128 output channels (8 acc tiles).
__global__ void __launch_bounds__(128) attn_main(const _Float16* __restrict__ fT,
                                                 const _Float16* __restrict__ gT,
                                                 const _Float16* __restrict__ vs,
                                                 const float* __restrict__ shift,
                                                 const float* __restrict__ x,
                                                 const float* __restrict__ gamma,
                                                 float* __restrict__ out) {
  __shared__ float ssh[N_];
  const int t   = threadIdx.x;
  const int l   = t & 31;
  const int w   = t >> 5;
  const int r16 = l & 15;
  const int h   = l >> 4;
  const int b   = blockIdx.y;
  const int m0  = (blockIdx.x * 4 + w) * 16;
  const float lg = gamma[0];

  for (int i = t; i < N_; i += 128) ssh[i] = shift[b * N_ + i];
  __syncthreads();

  // B operand for g is constant over the whole n loop: column m0+r16.
  const v16h g = *(const v16h*)(gT + ((size_t)(b * N_ + m0 + r16)) * CK_ + 16 * h);

  v8f acc[8];
  v8f zz = {};
#pragma unroll
  for (int ct = 0; ct < 8; ++ct) acc[ct] = zz;

  for (int nb = 0; nb < N_; nb += 32) {
    // Recompute s for rows nb..nb+31 against cols m0..m0+15 (two 16x16 tiles).
    const _Float16* fb = fT + ((size_t)(b * N_ + nb + r16)) * CK_ + 16 * h;
    v16h a1 = load_a16(fb);
    v16h a2 = load_a16(fb + 16 * CK_);
    v8f z = {};
    v8f s1 = __builtin_amdgcn_wmma_f32_16x16x32_f16(false, a1, false, g, (short)0, z, false, false);
    v8f s2 = __builtin_amdgcn_wmma_f32_16x16x32_f16(false, a2, false, g, (short)0, z, false, false);

    // Per-row softmax constants: 8 consecutive floats per tile-half (32B aligned)
    float sh1[8], sh2[8];
    *(float4*)&sh1[0] = *(const float4*)&ssh[nb + 8 * h];
    *(float4*)&sh1[4] = *(const float4*)&ssh[nb + 8 * h + 4];
    *(float4*)&sh2[0] = *(const float4*)&ssh[nb + 16 + 8 * h];
    *(float4*)&sh2[4] = *(const float4*)&ssh[nb + 16 + 8 * h + 4];

    // p = exp(s - shift_n): softmax numerator with denominator folded in.
    float e1[8], e2[8];
#pragma unroll
    for (int r = 0; r < 8; ++r) {
      e1[r] = __expf(s1[r] - sh1[r]);
      e2[r] = __expf(s2[r] - sh2[r]);
    }

    // Repack D-layout (lane=col, vgpr=row r+8h) into this code's contiguous-K f16 B-operand layout
    // (lane half 0 holds K=0..15, half 1 holds K=16..31). Convert to packed
    // f16 pairs first, then one 32-bit half-swap shuffle per pair.
    union { v16h v; unsigned int u[8]; } P;
#pragma unroll
    for (int j = 0; j < 4; ++j) {
      unsigned int pk1 = packh2(e1[2 * j], e1[2 * j + 1]);
      unsigned int pk2 = packh2(e2[2 * j], e2[2 * j + 1]);
      unsigned int o1 = (unsigned int)__shfl_xor((int)pk1, 16, 32);
      unsigned int o2 = (unsigned int)__shfl_xor((int)pk2, 16, 32);
      P.u[j]     = h ? o2 : pk1;   // elements 2j,2j+1   : K = 16h + 2j
      P.u[4 + j] = h ? pk2 : o1;   // elements 8+2j,9+2j : K = 16h + 8 + 2j
    }
    v16h p = P.v;

    // acc[c-tile] += v_tile(16c x 32n) @ p(32n x 16m)
    const _Float16* vb = vs + (size_t)b * C_ * N_ + (size_t)r16 * N_ + nb + 16 * h;
#pragma unroll
    for (int ct = 0; ct < 8; ++ct) {
      v16h av = load_a16(vb + (size_t)ct * 16 * N_);
      acc[ct] = __builtin_amdgcn_wmma_f32_16x16x32_f16(false, av, false, p, (short)0,
                                                       acc[ct], false, false);
    }
  }

  // Epilogue: out = gamma * sa + x. D layout: lane=col(r16), rows r+8h.
#pragma unroll
  for (int ct = 0; ct < 8; ++ct) {
#pragma unroll
    for (int r = 0; r < 8; ++r) {
      int c = ct * 16 + r + 8 * h;
      size_t o = ((size_t)(b * C_ + c)) * N_ + m0 + r16;
      out[o] = lg * acc[ct][r] + x[o];
    }
  }
}

extern "C" void kernel_launch(void* const* d_in, const int* in_sizes, int n_in,
                              void* d_out, int out_size, void* d_ws, size_t ws_size,
                              hipStream_t stream) {
  const float* x     = (const float*)d_in[0];
  const float* Wf    = (const float*)d_in[1];
  const float* bf    = (const float*)d_in[2];
  const float* Wg    = (const float*)d_in[3];
  const float* bg    = (const float*)d_in[4];
  const float* Wh    = (const float*)d_in[5];
  const float* bh    = (const float*)d_in[6];
  const float* gamma = (const float*)d_in[7];
  float* out = (float*)d_out;

  char* ws = (char*)d_ws;
  _Float16* fT    = (_Float16*)(ws);
  _Float16* gT    = (_Float16*)(ws + (1u << 20));
  _Float16* vsbuf = (_Float16*)(ws + (2u << 20));
  float*    shift = (float*)   (ws + (6u << 20));

  proj_fg<<<(B_ * N_ * CK_) / 256, 256, 0, stream>>>(x, Wf, bf, Wg, bg, fT, gT);
  proj_v <<<(B_ * C_ * N_) / 256, 256, 0, stream>>>(x, Wh, bh, vsbuf);
  attn_stats<<<dim3(N_ / 64, B_), 128, 0, stream>>>(fT, gT, shift);
  attn_main <<<dim3(N_ / 64, B_), 128, 0, stream>>>(fT, gT, vsbuf, shift, x, gamma, out);
}
